// BinaryTreeLSTMCell_81320910782709
// MI455X (gfx1250) — matrix operand order repaired and checked by reading, not compiled, not run on hardware
//
#include <hip/hip_runtime.h>
#include <hip/hip_bf16.h>

// ---------------------------------------------------------------------------
// Tree-LSTM on MI455X (gfx1250, wave32, WMMA).
//   B=64, L=256, H=1024, 8 levels (128 -> 1 merges per batch element).
// Each level: G = [hl|hr](M,2H) @ Wcat(2H,5H) + bcat, fused gate epilogue.
// bf16 WMMA + f32 accum, K-step 64, double-buffered LDS software pipeline,
// weights staged with global_load_async_to_lds_b128 (ASYNCcnt) when available.
// ---------------------------------------------------------------------------

#define HDIM   1024
#define LSEQ   256
#define BATCH  64
#define FOREST 255
#define XPITCH 72     // shorts per LDS row (144 B: 16B-aligned, bank-conflict free)

#ifndef __has_builtin
#define __has_builtin(x) 0
#endif
#if __has_builtin(__builtin_amdgcn_global_load_async_to_lds_b128)
#define USE_ASYNC 1
#else
#define USE_ASYNC 0
#endif

typedef __attribute__((ext_vector_type(16))) __bf16 v16bf;
typedef __attribute__((ext_vector_type(8)))  float  v8f;
typedef __attribute__((ext_vector_type(4)))  int    v4i;

union Frag {
    uint4 u[2];
    v16bf v;
};

static __device__ __forceinline__ unsigned short f2bf_bits(float f) {
    unsigned u = __float_as_uint(f);
    unsigned r = u + 0x7FFFu + ((u >> 16) & 1u);   // round-to-nearest-even
    return (unsigned short)(r >> 16);
}
static __device__ __forceinline__ unsigned pack2(float a, float b) {
    return (unsigned)f2bf_bits(a) | ((unsigned)f2bf_bits(b) << 16);
}
static __device__ __forceinline__ float fsig(float x) {
    return __builtin_amdgcn_rcpf(1.0f + __expf(-x));
}
static __device__ __forceinline__ float ftanh(float x) {
    return 2.0f * __builtin_amdgcn_rcpf(1.0f + __expf(-2.0f * x)) - 1.0f;
}

// ---------------------------------------------------------------------------
// Prep: Wcat^T bf16, (5H gate-cols) x (2H K).  Gate col n: [0,3H)=iock,
// [3H,4H)=fl, [4H,5H)=fr ; K<H from W, K>=H from U.
// ---------------------------------------------------------------------------
__global__ __launch_bounds__(256) void prep_w(
    const float* __restrict__ Wi, const float* __restrict__ Ui,
    const float* __restrict__ Wfl, const float* __restrict__ Ufl,
    const float* __restrict__ Wfr, const float* __restrict__ Ufr,
    unsigned short* __restrict__ Wt)
{
    const int idx = blockIdx.x * 256 + threadIdx.x;   // 5H*256 threads
    const int n   = idx >> 8;                         // gate column 0..5119
    const int k8  = (idx & 255) << 3;                 // 8 K per thread

    const float *Ws, *Us; int col, stride;
    if (n < 3 * HDIM)      { Ws = Wi;  Us = Ui;  col = n;            stride = 3 * HDIM; }
    else if (n < 4 * HDIM) { Ws = Wfl; Us = Ufl; col = n - 3 * HDIM; stride = HDIM; }
    else                   { Ws = Wfr; Us = Ufr; col = n - 4 * HDIM; stride = HDIM; }

    uint4 v;
    unsigned* vp = (unsigned*)&v;
#pragma unroll
    for (int j = 0; j < 8; j += 2) {
        const int k0 = k8 + j, k1 = k0 + 1;
        const float f0 = (k0 < HDIM) ? Ws[(size_t)k0 * stride + col]
                                     : Us[(size_t)(k0 - HDIM) * stride + col];
        const float f1 = (k1 < HDIM) ? Ws[(size_t)k1 * stride + col]
                                     : Us[(size_t)(k1 - HDIM) * stride + col];
        vp[j >> 1] = pack2(f0, f1);
    }
    *(uint4*)&Wt[(size_t)n * (2 * HDIM) + k8] = v;
}

__global__ __launch_bounds__(256) void prep_b(
    const float* __restrict__ bi, const float* __restrict__ bfl,
    const float* __restrict__ bfr, float* __restrict__ bcat)
{
    const int n = blockIdx.x * 256 + threadIdx.x;
    float v;
    if (n < 3 * HDIM)      v = bi[n];
    else if (n < 4 * HDIM) v = bfl[n - 3 * HDIM];
    else                   v = bfr[n - 4 * HDIM];
    bcat[n] = v;
}

// ---------------------------------------------------------------------------
// Pipeline stages for the level kernel
// ---------------------------------------------------------------------------
struct XStage { float4 f[8]; };     // 32 floats = one (row, 32-K) strip

static __device__ __forceinline__ void x_issue(
    XStage& st, int tid, int m0, int kc, int M, int halfn, int in_off,
    const int* __restrict__ tokens, const float* __restrict__ emb,
    const float* __restrict__ h_in)
{
    const int rl = tid >> 1;
    const int kh = (tid & 1) << 5;            // 0 or 32
    const int m  = m0 + rl;
    if (m < M) {
        const int b = m / halfn;
        const int p = m - b * halfn;
        const int kk = kc + kh;
        const float* src;
        if (h_in == nullptr) {                // level 0: gather from embedding
            const int which = (kk >= HDIM) ? 1 : 0;
            const int tok = tokens[b * LSEQ + 2 * p + which];
            src = emb + (size_t)tok * HDIM + (kk - which * HDIM);
        } else {
            src = h_in + (size_t)b * (FOREST * HDIM)
                       + (size_t)(in_off + 2 * p) * HDIM + kk;
        }
        const float4* s4 = (const float4*)src;
#pragma unroll
        for (int i = 0; i < 8; ++i) st.f[i] = s4[i];
    } else {
        const float4 z = {0.f, 0.f, 0.f, 0.f};
#pragma unroll
        for (int i = 0; i < 8; ++i) st.f[i] = z;
    }
}

static __device__ __forceinline__ void x_commit(
    const XStage& st, int tid, unsigned short* bufX)
{
    const int rl = tid >> 1;
    const int kh = (tid & 1) << 5;
    unsigned short* dst = &bufX[rl * XPITCH + kh];
#pragma unroll
    for (int q = 0; q < 4; ++q) {
        uint4 pk;
        pk.x = pack2(st.f[2*q].x,   st.f[2*q].y);
        pk.y = pack2(st.f[2*q].z,   st.f[2*q].w);
        pk.z = pack2(st.f[2*q+1].x, st.f[2*q+1].y);
        pk.w = pack2(st.f[2*q+1].z, st.f[2*q+1].w);
        *(uint4*)(dst + 8 * q) = pk;
    }
}

#if USE_ASYNC
// global_load_async_to_lds_b128: (int4 AS1* gsrc, int4 AS3* ldst, imm off, imm cpol)
static __device__ __forceinline__ void async_copy16(
    const unsigned short* src, unsigned short* dst)
{
    __builtin_amdgcn_global_load_async_to_lds_b128(
        (__attribute__((address_space(1))) v4i*)(v4i*)(size_t)src,
        (__attribute__((address_space(3))) v4i*)(v4i*)dst,
        0, 0);
}

static __device__ __forceinline__ void w_issue_async(
    int tid, int n0, int kc, const unsigned short* __restrict__ Wt,
    unsigned short* bufW)
{
    // 5 gates x 32 rows x 128B ; 1280 16B-chunks / 256 threads = 5 each
#pragma unroll
    for (int j = 0; j < 5; ++j) {
        const int c   = tid + 256 * j;
        const int row = c >> 3;               // 0..159
        const int sub = (c & 7) << 3;         // 0..56 shorts
        const int g   = row >> 5;
        const int r   = row & 31;
        const unsigned short* src =
            Wt + (size_t)(g * HDIM + n0 + r) * (2 * HDIM) + kc + sub;
        unsigned short* dst = &bufW[(g * 32 + r) * XPITCH + sub];
        async_copy16(src, dst);
    }
}
#else
struct WStage { uint4 w[5]; };

static __device__ __forceinline__ void w_issue_sync(
    WStage& st, int tid, int n0, int kc, const unsigned short* __restrict__ Wt)
{
#pragma unroll
    for (int j = 0; j < 5; ++j) {
        const int c   = tid + 256 * j;
        const int row = c >> 3;
        const int sub = (c & 7) << 3;
        const int g   = row >> 5;
        const int r   = row & 31;
        st.w[j] = *(const uint4*)(Wt + (size_t)(g * HDIM + n0 + r) * (2 * HDIM) + kc + sub);
    }
}

static __device__ __forceinline__ void w_commit_sync(
    const WStage& st, int tid, unsigned short* bufW)
{
#pragma unroll
    for (int j = 0; j < 5; ++j) {
        const int c   = tid + 256 * j;
        const int row = c >> 3;
        const int sub = (c & 7) << 3;
        const int g   = row >> 5;
        const int r   = row & 31;
        *(uint4*)(&bufW[(g * 32 + r) * XPITCH + sub]) = st.w[j];
    }
}
#endif

// 20 WMMAs over a 64-wide K slab held in LDS
static __device__ __forceinline__ void compute64(
    v8f acc[5][2], const unsigned short* bufX, const unsigned short* bufW,
    int wave, int lane)
{
#pragma unroll
    for (int s = 0; s < 2; ++s) {
        Frag fa;   // A 16x32: lane row = lane&15, K = kb+[0,16)
        {
            const int row = (wave << 4) + (lane & 15);
            const int kb  = (lane >> 4) << 4;          // 0 or 16
            const unsigned short* p = &bufX[row * XPITCH + s * 32 + kb];
            fa.u[0] = *(const uint4*)p;
            fa.u[1] = *(const uint4*)(p + 8);
        }
#pragma unroll
        for (int g = 0; g < 5; ++g) {
#pragma unroll
            for (int t = 0; t < 2; ++t) {
                Frag fb;  // B 32x16: lane col = lane&15, K = 16*(lane>=16)+[0,16)
                const int nl  = (t << 4) + (lane & 15);
                const int kb2 = (lane >> 4) << 4;      // 0 or 16
                const unsigned short* q = &bufW[(g * 32 + nl) * XPITCH + s * 32 + kb2];
                fb.u[0] = ((const uint4*)q)[0];
                fb.u[1] = ((const uint4*)q)[1];
                acc[g][t] = __builtin_amdgcn_wmma_f32_16x16x32_bf16(
                    false, fa.v, false, fb.v, (short)0, acc[g][t], false, false);
            }
        }
    }
}

// ---------------------------------------------------------------------------
// One tree level. WG = 8 waves, tile = 128 rows x 32 gate-cols x 5 gates.
// ---------------------------------------------------------------------------
__global__ __launch_bounds__(256) void tree_level_kernel(
    const int*   __restrict__ tokens,
    const float* __restrict__ emb,
    const float* __restrict__ h_in,          // null => level-0 embedding gather
    const float* __restrict__ c_in,          // null => c = 0
    const unsigned short* __restrict__ Wt,   // (5H, 2H) bf16
    const float* __restrict__ bcat,          // (5H)
    float* __restrict__ out,                 // forest (B,255,H) + h/c root tails
    float* __restrict__ c_out,               // (B, halfn, H)
    int M, int halfn, int in_off, int out_off, int is_last)
{
    const int tid  = threadIdx.x;
    const int lane = tid & 31;
    const int wave = tid >> 5;
    const int m0   = blockIdx.x * 128;
    const int n0   = blockIdx.y * 32;

    __shared__ unsigned short ldsX[2][128 * XPITCH];      // 2 x 18.0 KB
    __shared__ unsigned short ldsW[2][5 * 32 * XPITCH];   // 2 x 22.5 KB

    v8f acc[5][2];
#pragma unroll
    for (int g = 0; g < 5; ++g)
#pragma unroll
        for (int t = 0; t < 2; ++t) {
            v8f z = {0.f, 0.f, 0.f, 0.f, 0.f, 0.f, 0.f, 0.f};
            acc[g][t] = z;
        }

    // ---- prologue: stage K slab 0 into buffer 0 ----
    {
        XStage xs0;
        x_issue(xs0, tid, m0, 0, M, halfn, in_off, tokens, emb, h_in);
#if USE_ASYNC
        w_issue_async(tid, n0, 0, Wt, &ldsW[0][0]);
#else
        WStage w0;
        w_issue_sync(w0, tid, n0, 0, Wt);
#endif
        x_commit(xs0, tid, &ldsX[0][0]);
#if USE_ASYNC
        asm volatile("s_wait_asynccnt 0x0" ::: "memory");
#else
        w_commit_sync(w0, tid, &ldsW[0][0]);
#endif
    }
    __syncthreads();

    // ---- pipelined main loop: prefetch slab kc+64 while computing slab kc ----
    for (int kc = 0; kc < 2 * HDIM; kc += 64) {
        const int  cur = (kc >> 6) & 1;
        const bool pf  = (kc + 64) < 2 * HDIM;

        XStage xs;
#if !USE_ASYNC
        WStage wst;
#endif
        if (pf) {
            x_issue(xs, tid, m0, kc + 64, M, halfn, in_off, tokens, emb, h_in);
#if USE_ASYNC
            w_issue_async(tid, n0, kc + 64, Wt, &ldsW[cur ^ 1][0]);
#else
            w_issue_sync(wst, tid, n0, kc + 64, Wt);
#endif
        }

        compute64(acc, &ldsX[cur][0], &ldsW[cur][0], wave, lane);

        if (pf) {
            x_commit(xs, tid, &ldsX[cur ^ 1][0]);
#if USE_ASYNC
            asm volatile("s_wait_asynccnt 0x0" ::: "memory");
#else
            w_commit_sync(wst, tid, &ldsW[cur ^ 1][0]);
#endif
        }
        __syncthreads();
    }

    // ---- fused gate epilogue: D element (M = r + 8*(lane>=16), N = lane&15) ----
#pragma unroll
    for (int t = 0; t < 2; ++t) {
        const int n   = n0 + (t << 4) + (lane & 15);
        const float bi  = bcat[n];
        const float bo  = bcat[HDIM + n];
        const float bck = bcat[2 * HDIM + n];
        const float bfl = bcat[3 * HDIM + n];
        const float bfr = bcat[4 * HDIM + n];
#pragma unroll
        for (int r = 0; r < 8; ++r) {
            const int m = m0 + (wave << 4) + r + ((lane >> 4) << 3);
            if (m >= M) continue;
            const int b = m / halfn;
            const int p = m - b * halfn;

            const float iv  = acc[0][t][r] + bi;
            const float ov  = acc[1][t][r] + bo;
            const float ckv = acc[2][t][r] + bck;
            const float flv = acc[3][t][r] + bfl;
            const float frv = acc[4][t][r] + bfr;

            float cn = fsig(iv) * ftanh(ckv);
            if (c_in) {
                const size_t cb = ((size_t)b * (halfn << 1) + (p << 1)) * HDIM + n;
                cn += fsig(flv) * c_in[cb] + fsig(frv) * c_in[cb + HDIM];
            }
            const float hn = fsig(ov) * ftanh(cn);

            c_out[((size_t)b * halfn + p) * HDIM + n] = cn;
            out[(size_t)b * (FOREST * HDIM) + (size_t)(out_off + p) * HDIM + n] = hn;
            if (is_last)
                out[(size_t)BATCH * FOREST * HDIM + (size_t)b * HDIM + n] = hn;
        }
    }
}

// ---------------------------------------------------------------------------
extern "C" void kernel_launch(void* const* d_in, const int* in_sizes, int n_in,
                              void* d_out, int out_size, void* d_ws, size_t ws_size,
                              hipStream_t stream)
{
    (void)in_sizes; (void)n_in; (void)out_size; (void)ws_size;

    const int*   tokens = (const int*)  d_in[0];
    const float* emb    = (const float*)d_in[1];
    const float* W_iock = (const float*)d_in[2];
    const float* b_iock = (const float*)d_in[3];
    const float* U_iock = (const float*)d_in[4];
    const float* W_fl   = (const float*)d_in[5];
    const float* b_fl   = (const float*)d_in[6];
    const float* U_fl   = (const float*)d_in[7];
    const float* W_fr   = (const float*)d_in[8];
    const float* b_fr   = (const float*)d_in[9];
    const float* U_fr   = (const float*)d_in[10];
    float* out = (float*)d_out;

    // ws: bf16 Wcat^T (20 MB) | bcat (32 KB pad) | c ping (33.5 MB) | c pong (16.8 MB)
    unsigned short* Wt = (unsigned short*)d_ws;
    const size_t wt_bytes = (size_t)5 * HDIM * 2 * HDIM * sizeof(unsigned short);
    float* bcat = (float*)((char*)d_ws + wt_bytes);
    float* cA   = (float*)((char*)d_ws + wt_bytes + 32768);
    float* cB   = cA + (size_t)BATCH * 128 * HDIM;

    prep_w<<<dim3(5 * HDIM), dim3(256), 0, stream>>>(W_iock, U_iock, W_fl, U_fl, W_fr, U_fr, Wt);
    prep_b<<<dim3(20),       dim3(256), 0, stream>>>(b_iock, b_fl, b_fr, bcat);

    static const int off[8] = {0, 128, 192, 224, 240, 248, 252, 254};
    const float* c_in = nullptr;
    for (int lvl = 0; lvl < 8; ++lvl) {
        const int halfn = 128 >> lvl;
        const int M     = BATCH * halfn;
        const int last  = (lvl == 7);
        float* c_out = last ? (out + (size_t)BATCH * FOREST * HDIM + (size_t)BATCH * HDIM)
                            : ((lvl & 1) ? cB : cA);
        const float* h_in = (lvl == 0) ? nullptr : out;
        const int in_off  = (lvl == 0) ? 0 : off[lvl - 1];

        dim3 grid((M + 127) / 128, HDIM / 32);
        tree_level_kernel<<<grid, dim3(256), 0, stream>>>(
            tokens, emb, h_in, c_in, Wt, bcat, out, c_out,
            M, halfn, in_off, off[lvl], last);
        c_in = c_out;
    }
}
